// DenseGraphAttentionLayer_19894288515591
// MI455X (gfx1250) — matrix operand order repaired and checked by reading, not compiled, not run on hardware
//
#include <hip/hip_runtime.h>

#define N_NODES 8192
#define F_IN    256
#define F_OUT   128
#define SLOPE   0.2f
#define NEG_INF -9000000000000000.0f
#define WPB     4        // waves per 16-row block (column split)
#define SEG     (N_NODES / WPB)   // 2048 columns per wave

typedef __attribute__((ext_vector_type(2)))  float  v2f;
typedef __attribute__((ext_vector_type(4)))  float  v4f;
typedef __attribute__((ext_vector_type(8)))  float  v8f;
typedef __attribute__((ext_vector_type(16))) __bf16 v16bf;

static __device__ __forceinline__ unsigned short f32_to_bf16(float f) {
  unsigned int u = __float_as_uint(f);
  u += 0x7FFFu + ((u >> 16) & 1u);   // round-to-nearest-even
  return (unsigned short)(u >> 16);
}

// ---------------------------------------------------------------------------
// Kernel 1: h = x @ W  (fp32 WMMA 16x16x4), f_src/f_dst reductions, and
// store h^T in bf16. One wave per 16-row stripe, 8 tiles across F_OUT.
// ---------------------------------------------------------------------------
__global__ void gat_hproj(const float* __restrict__ x,
                          const float* __restrict__ W,
                          const float* __restrict__ a_src,
                          const float* __restrict__ a_dst,
                          unsigned short* __restrict__ hT,   // [F_OUT][N_NODES] bf16
                          float* __restrict__ f_src,
                          float* __restrict__ f_dst) {
  const int lane = threadIdx.x & 31;
  const int wave = (int)((blockIdx.x * blockDim.x + threadIdx.x) >> 5);
  const int rb   = wave;              // 16-row block index [0,512)
  const int lm   = lane & 15;
  const int kh   = lane >> 4;
  const int row  = rb * 16 + lm;

  v8f zero = {};
  v8f c[8];
#pragma unroll
  for (int t = 0; t < 8; ++t) c[t] = zero;

  const float* xrow = x + (size_t)row * F_IN;

  for (int k = 0; k < F_IN; k += 4) {
    // A-matrix 16x4 f32: lanes 0-15 -> K=0,1 ; lanes 16-31 -> K=2,3
    v2f a;
    a.x = xrow[k + kh * 2 + 0];
    a.y = xrow[k + kh * 2 + 1];
#pragma unroll
    for (int t = 0; t < 8; ++t) {
      const int col = t * 16 + lm;
      v2f b;
      b.x = W[(size_t)(k + kh * 2 + 0) * F_OUT + col];
      b.y = W[(size_t)(k + kh * 2 + 1) * F_OUT + col];
      c[t] = __builtin_amdgcn_wmma_f32_16x16x4_f32(false, a, false, b,
                                                   (short)0, c[t], false, false);
    }
  }

  // Store h^T bf16: C element (t, v) -> row (rb*16 + v + 8*kh), col (16t + lm)
#pragma unroll
  for (int t = 0; t < 8; ++t) {
#pragma unroll
    for (int v = 0; v < 8; ++v) {
      const int r   = rb * 16 + v + 8 * kh;
      const int col = t * 16 + lm;
      hT[(size_t)col * N_NODES + r] = f32_to_bf16(c[t][v]);
    }
  }

  // f_src / f_dst reductions across the 16 lanes of each half.
  float ps[8], pd[8];
#pragma unroll
  for (int v = 0; v < 8; ++v) { ps[v] = 0.f; pd[v] = 0.f; }
#pragma unroll
  for (int t = 0; t < 8; ++t) {
    const float as = a_src[t * 16 + lm];
    const float ad = a_dst[t * 16 + lm];
#pragma unroll
    for (int v = 0; v < 8; ++v) {
      ps[v] += c[t][v] * as;
      pd[v] += c[t][v] * ad;
    }
  }
#pragma unroll
  for (int off = 1; off < 16; off <<= 1) {
#pragma unroll
    for (int v = 0; v < 8; ++v) {
      ps[v] += __shfl_xor(ps[v], off, 32);
      pd[v] += __shfl_xor(pd[v], off, 32);
    }
  }
  if (lm == 0) {
#pragma unroll
    for (int v = 0; v < 8; ++v) {
      const int r = rb * 16 + kh * 8 + v;
      f_src[r] = ps[v];
      f_dst[r] = pd[v];
    }
  }
}

// ---------------------------------------------------------------------------
// Kernel 2: fused masked-leaky-relu + online softmax + alpha@h.
// One workgroup (4 waves) per 16-row block; each wave streams a 2048-column
// segment of adj with NT hints + prefetch, accumulating a partial flash
// state (m, s, acc) with bf16 WMMA. LDS log-sum-exp merge at the end.
// ---------------------------------------------------------------------------
__global__ void gat_attn(const float* __restrict__ adj,
                         const unsigned short* __restrict__ hT,
                         const float* __restrict__ f_src,
                         const float* __restrict__ f_dst,
                         float* __restrict__ out) {
  __shared__ float ls_acc[WPB][16 * F_OUT];   // 32 KB
  __shared__ float ls_m[WPB][16];
  __shared__ float ls_s[WPB][16];

  const int lane = threadIdx.x & 31;
  const int w    = threadIdx.x >> 5;          // wave in block [0,WPB)
  const int rb   = blockIdx.x;                // 16-row block [0,512)
  const int lm   = lane & 15;
  const int kh   = lane >> 4;
  const int row  = rb * 16 + lm;

  const float fs = f_src[row];

  float m = NEG_INF;
  float s = 0.f;
  v8f zero = {};
  v8f acc[8];
#pragma unroll
  for (int t = 0; t < 8; ++t) acc[t] = zero;

  const float* adjrow = adj + (size_t)row * N_NODES;
  const int jstart = w * SEG;
  const int jend   = jstart + SEG;

  for (int j = jstart; j < jend; j += 32) {
    // Columns this lane covers (matches 16-bit A-matrix 16x32 layout):
    // lanes 0-15: K = 0..7 and 16..23 ; lanes 16-31: K = 8..15 and 24..31
    const int cA = j + kh * 8;
    const int cB = j + 16 + kh * 8;

    // Prefetch adj 8 steps (256 columns) ahead within this segment.
    if (j + 256 < jend) {
      __builtin_prefetch(adjrow + cA + 256, 0, 0);
      __builtin_prefetch(adjrow + cB + 256, 0, 0);
    }

    __align__(16) float fd[16];
    __align__(16) float ad[16];
    *(v4f*)(fd + 0)  = *(const v4f*)(f_dst + cA);
    *(v4f*)(fd + 4)  = *(const v4f*)(f_dst + cA + 4);
    *(v4f*)(fd + 8)  = *(const v4f*)(f_dst + cB);
    *(v4f*)(fd + 12) = *(const v4f*)(f_dst + cB + 4);
    // adj is single-touch (256 MB > L2): non-temporal stream.
    *(v4f*)(ad + 0)  = __builtin_nontemporal_load((const v4f*)(adjrow + cA));
    *(v4f*)(ad + 4)  = __builtin_nontemporal_load((const v4f*)(adjrow + cA + 4));
    *(v4f*)(ad + 8)  = __builtin_nontemporal_load((const v4f*)(adjrow + cB));
    *(v4f*)(ad + 12) = __builtin_nontemporal_load((const v4f*)(adjrow + cB + 4));

    float e[16];
    float tmax = NEG_INF;
#pragma unroll
    for (int i = 0; i < 16; ++i) {
      float v = fs + fd[i];
      v = (v >= 0.f) ? v : SLOPE * v;            // leaky relu
      v = (ad[i] > 0.f) ? v : NEG_INF;           // adjacency mask
      e[i] = v;
      tmax = fmaxf(tmax, v);
    }
    tmax = fmaxf(tmax, __shfl_xor(tmax, 16, 32)); // pair the two lanes per row

    const float m_new = fmaxf(m, tmax);
    const float scale = __expf(m - m_new);

    float p[16];
    float tsum = 0.f;
#pragma unroll
    for (int i = 0; i < 16; ++i) {
      p[i] = __expf(e[i] - m_new);
      tsum += p[i];
    }
    tsum += __shfl_xor(tsum, 16, 32);
    s = s * scale + tsum;
    m = m_new;

    // Pack P tile (A matrix) to bf16.
    union { v16bf v; unsigned short u[16]; } A;
#pragma unroll
    for (int i = 0; i < 16; ++i) A.u[i] = f32_to_bf16(p[i]);

    // Rescale accumulators: C VGPR v holds row (v + 8*kh); scale lives in
    // lane (v + 8*kh) (duplicated at +16).
    float sclv[8];
#pragma unroll
    for (int v = 0; v < 8; ++v) sclv[v] = __shfl(scale, v + 8 * kh, 32);
#pragma unroll
    for (int t = 0; t < 8; ++t) {
#pragma unroll
      for (int v = 0; v < 8; ++v) acc[t][v] *= sclv[v];
    }

    // B tiles: 32x16 bf16; lane holds col N = 16t+lm, K = kh*8 + 0..7 and
    // 16 + kh*8 + 0..7 in h^T -> two 16B loads per tile (L2-resident).
#pragma unroll
    for (int t = 0; t < 8; ++t) {
      union { v16bf v; uint4 q[2]; } B;
      const unsigned short* bp =
          hT + (size_t)(t * 16 + lm) * N_NODES + (j + kh * 8);
      B.q[0] = *(const uint4*)(bp);
      B.q[1] = *(const uint4*)(bp + 16);
      acc[t] = __builtin_amdgcn_wmma_f32_16x16x32_bf16(false, A.v, false, B.v,
                                                       (short)0, acc[t],
                                                       false, false);
    }
  }

  // ---- publish partial flash state to LDS ----
#pragma unroll
  for (int t = 0; t < 8; ++t) {
#pragma unroll
    for (int v = 0; v < 8; ++v) {
      ls_acc[w][(v + 8 * kh) * F_OUT + t * 16 + lm] = acc[t][v];
    }
  }
  if (kh == 0) {
    ls_m[w][lm] = m;
    ls_s[w][lm] = s;
  }
  __syncthreads();

  // ---- log-sum-exp merge across the WPB column segments ----
  // 128 threads: thread handles row r = tid>>3, columns [(tid&7)*16, +16).
  const int tid = threadIdx.x;
  const int r   = tid >> 3;
  const int c0  = (tid & 7) * 16;

  float mw[WPB];
  float M = NEG_INF;
#pragma unroll
  for (int q = 0; q < WPB; ++q) {
    mw[q] = ls_m[q][r];
    M = fmaxf(M, mw[q]);
  }
  float coef[WPB];
  float den = 0.f;
#pragma unroll
  for (int q = 0; q < WPB; ++q) {
    coef[q] = __expf(mw[q] - M);
    den += ls_s[q][r] * coef[q];
  }
  const float dinv = 1.0f / den;

  float* orow = out + (size_t)(rb * 16 + r) * F_OUT + c0;
#pragma unroll
  for (int i = 0; i < 16; ++i) {
    float v = 0.f;
#pragma unroll
    for (int q = 0; q < WPB; ++q) {
      v += ls_acc[q][r * F_OUT + c0 + i] * coef[q];
    }
    orow[i] = v * dinv;
  }
}

// ---------------------------------------------------------------------------
extern "C" void kernel_launch(void* const* d_in, const int* in_sizes, int n_in,
                              void* d_out, int out_size, void* d_ws, size_t ws_size,
                              hipStream_t stream) {
  const float* x     = (const float*)d_in[0];
  const float* adj   = (const float*)d_in[1];
  const float* W     = (const float*)d_in[2];
  const float* a_src = (const float*)d_in[3];
  const float* a_dst = (const float*)d_in[4];
  float* out = (float*)d_out;

  unsigned short* hT = (unsigned short*)d_ws;                       // 2 MB
  float* f_src = (float*)((char*)d_ws + (size_t)F_OUT * N_NODES * 2);
  float* f_dst = f_src + N_NODES;

  // Kernel 1: 512 waves (16-row stripes), 4 waves per workgroup.
  gat_hproj<<<128, 128, 0, stream>>>(x, W, a_src, a_dst, hT, f_src, f_dst);
  // Kernel 2: one workgroup per 16-row block, 4 column-segment waves each.
  gat_attn <<<512, 32 * WPB, 0, stream>>>(adj, hT, f_src, f_dst, out);
}
